// MusicTransformer_33354716021081
// MI455X (gfx1250) — matrix operand order repaired and checked by reading, not compiled, not run on hardware
//
#include <hip/hip_runtime.h>

// ---------------------------------------------------------------------------
// MusicTransformer forward on gfx1250 (MI455X).
// bf16 WMMA (16x16x32, fp32 accum) for all GEMMs; fp32 softmax/LayerNorm.
// GEMM staging: A tiles via GLOBAL_LOAD_ASYNC_TO_LDS_B128 (ASYNCcnt),
// B (weight) tiles via the Tensor Data Mover (TENSOR_LOAD_TO_LDS, TENSORcnt)
// with hardware LDS row padding; both double-buffered under the WMMA stream.
// Dims: V=512 D=512 H=8 L=6 F=2048 B=8 S=1024 DK=64, tokens=8192
// ---------------------------------------------------------------------------

#define Vv   512
#define Dd   512
#define Hh   8
#define Ll   6
#define Ff   2048
#define Bb   8
#define Ss   1024
#define DKk  64
#define MTOK (Bb * Ss)      // 8192
#define QKVN (3 * Dd)       // 1536
#define SQRTD 22.62741699796952f
#define NEGBIG -1.0e9f
#define PADW 40             // LDS row stride (elems): 32 data + 8 pad

typedef __bf16        v16bf __attribute__((ext_vector_type(16)));
typedef float         v8f   __attribute__((ext_vector_type(8)));
typedef unsigned int  u32x4 __attribute__((ext_vector_type(4)));
typedef unsigned int  su4   __attribute__((ext_vector_type(4)));
typedef unsigned int  su8   __attribute__((ext_vector_type(8)));
typedef unsigned short u16;

// fp32 -> bf16, round-to-nearest-even
__device__ __forceinline__ u16 f2b(float f) {
    unsigned u = __builtin_bit_cast(unsigned, f);
    u += 0x7fffu + ((u >> 16) & 1u);
    return (u16)(u >> 16);
}

union FragB { v16bf bf; u32x4 u[2]; u16 us[16]; };

__device__ __forceinline__ v8f wmma_bf16(v16bf a, v16bf b, v8f c) {
    return __builtin_amdgcn_wmma_f32_16x16x32_bf16(false, a, false, b, (short)0, c,
                                                   false, false);
}

__device__ __forceinline__ u32x4 ld16(const u16* p) { return *(const u32x4*)p; }

// Generic pointer -> wave-relative LDS byte offset (aperture low 32 bits).
__device__ __forceinline__ unsigned ldsoff(const void* p) {
    return (unsigned)(size_t)p;
}

// One 16-byte async DMA chunk: global -> LDS, tracked by ASYNCcnt.
__device__ __forceinline__ void async_b128(unsigned lds_addr, const u16* g) {
    asm volatile("global_load_async_to_lds_b128 %0, %1, off"
                 :: "v"(lds_addr), "v"((unsigned long long)(size_t)g)
                 : "memory");
}

__device__ __forceinline__ void wait_async0() {
    asm volatile("s_wait_asynccnt 0x0" ::: "memory");
}

// ---------------------------------------------------------------------------
// Tensor Data Mover: load a 2D tile (tile_rows x 32 bf16, row stride Kelems)
// into LDS with hardware padding of 4 DWORDs after every 16 DWORDs, producing
// the PADW=40-halfword row layout the WMMA fragment readers expect.
//   D# group0: count=1 | lds_addr | global_addr(57b) | type=2
//   D# group1: data_size=2B, pad_enable, pad_interval=3 (16 DW), pad_amount=3
//              (4 DW), tensor_dim0=K, tensor_dim1=rows, tile 32 x tile_rows,
//              tensor_dim0_stride=K
// Issued once per executing wave (EXEC-independent) -> call from wave 0 only;
// completion via TENSORcnt (s_wait_tensorcnt).
// ---------------------------------------------------------------------------
__device__ __forceinline__ void tdm_load_2d(unsigned lds_addr, const u16* g,
                                            unsigned tile_rows,
                                            unsigned tensor_rows,
                                            unsigned Kelems) {
    unsigned long long ga = (unsigned long long)(size_t)g;
    su4 g0;
    g0[0] = 1u;                                   // count=1, user mode
    g0[1] = lds_addr;                             // lds_addr[31:0]
    g0[2] = (unsigned)ga;                         // global_addr[31:0]
    g0[3] = (unsigned)(ga >> 32) | (2u << 30);    // global_addr[56:32] | type=2
    su8 g1;
    g1[0] = (1u << 16)      // data_size = 2 bytes
          | (1u << 20)      // pad_enable
          | (3u << 22)      // pad_interval: 16 DWORDs
          | (3u << 25);     // pad_amount:   4 DWORDs
    g1[1] = Kelems << 16;                         // tensor_dim0[15:0]
    g1[2] = (Kelems >> 16) | (tensor_rows << 16); // td0[31:16] | td1[15:0]
    g1[3] = (tensor_rows >> 16) | (32u << 16);    // td1[31:16] | tile_dim0=32
    g1[4] = tile_rows;                            // tile_dim1 (tile_dim2=0)
    g1[5] = Kelems;                               // tensor_dim0_stride[31:0]
    g1[6] = 0u;                                   // stride hi, dim1_stride lo
    g1[7] = 0u;
    asm volatile("tensor_load_to_lds %0, %1" :: "s"(g0), "s"(g1) : "memory");
}

// ---------------------------------------------------------------------------
// Weight convert + transpose: src f32 [batch,K,N] -> dst bf16 at
// dst[l*layerStride + base + n*K + k]  ([N,K] per layer, relocatable so
// Wq/Wk/Wv pack into one fused QKV weight buffer).
// ---------------------------------------------------------------------------
__global__ void convT_kernel(const float* __restrict__ src, u16* __restrict__ dst,
                             int K, int N, long total, long layerStride, long base) {
    long idx = (long)blockIdx.x * 256 + threadIdx.x;
    if (idx >= total) return;
    long per = (long)K * N;
    long l   = idx / per;
    long rem = idx - l * per;
    int  k   = (int)(rem / N);
    int  n   = (int)(rem - (long)k * N);
    dst[l * layerStride + base + (long)n * K + k] = f2b(src[idx]);
}

// Concatenate per-layer QKV biases: dst[l*1536 + {0:q,512:k,1024:v} + d]
__global__ void biascat_kernel(const float* __restrict__ bq,
                               const float* __restrict__ bk,
                               const float* __restrict__ bv,
                               float* __restrict__ dst) {
    int idx = blockIdx.x * 256 + threadIdx.x;
    if (idx >= Ll * QKVN) return;
    int l = idx / QKVN;
    int r = idx - l * QKVN;
    int w = r >> 9;
    int d = r & (Dd - 1);
    const float* s = (w == 0) ? bq : (w == 1) ? bk : bv;
    dst[idx] = s[l * Dd + d];
}

// ---------------------------------------------------------------------------
// Embedding: h = emb[ids] * sqrt(D) + pe[s]; writes f32 and bf16 copies.
// ---------------------------------------------------------------------------
__global__ __launch_bounds__(256)
void embed_kernel(const int* __restrict__ ids, const float* __restrict__ emb,
                  const float* __restrict__ pe, float* __restrict__ Hf,
                  u16* __restrict__ Hb) {
    int t   = blockIdx.x;
    int tid = threadIdx.x;
    int id  = ids[t];
    int s   = t & (Ss - 1);
    for (int d = tid; d < Dd; d += 256) {
        float v = emb[(size_t)id * Dd + d] * SQRTD + pe[(size_t)s * Dd + d];
        Hf[(size_t)t * Dd + d] = v;
        Hb[(size_t)t * Dd + d] = f2b(v);
    }
}

// ---------------------------------------------------------------------------
// V transpose out of the fused qkv buffer: qkv[:, 1024+c] -> Vt [B,H,DK,S]
// ---------------------------------------------------------------------------
__global__ void vtrans_kernel(const u16* __restrict__ QKV, u16* __restrict__ Vt) {
    size_t idx = (size_t)blockIdx.x * 256 + threadIdx.x;   // over MTOK*D
    int t  = (int)(idx >> 9);
    int c  = (int)(idx & 511);
    int h  = c >> 6;
    int dk = c & 63;
    int b  = t >> 10;
    int s  = t & 1023;
    Vt[(((size_t)(b * Hh + h)) * DKk + dk) * Ss + s] =
        QKV[(size_t)t * QKVN + 2 * Dd + c];
}

// ---------------------------------------------------------------------------
// GEMM: C[M,N] = act(A_bf16[M,K] @ Wt_bf16[N,K]^T + bias[N])
// 256 threads (8 waves), block tile 128x64, wave tile 32x32 (2x2 WMMA frags).
// K-tiles of 32 double-buffered in LDS: A via async DMA chunks, B via TDM.
// ---------------------------------------------------------------------------
__device__ __forceinline__ void stage_A(const u16* __restrict__ A, int K,
                                        int mrow0, int k0, u16* As, int tid) {
    for (int i = 0; i < 2; ++i) {     // 128 rows * 4 chunks = 512; 2/thread
        int c   = tid + i * 256;
        int row = c >> 2;
        int seg = c & 3;
        async_b128(ldsoff(As + row * PADW + seg * 8),
                   A + (size_t)(mrow0 + row) * K + k0 + seg * 8);
    }
}

template <bool OUT_BF16, bool RELU>
__global__ __launch_bounds__(256)
void gemm_kernel(const u16* __restrict__ A, const u16* __restrict__ Wt,
                 const float* __restrict__ bias, float* __restrict__ Cf,
                 u16* __restrict__ Cb, int M, int N, int K) {
    __shared__ u16 lds[2][(128 + 64) * PADW];   // 2 x 15 KB

    const int tid  = threadIdx.x;
    const int lane = tid & 31;
    const int wave = tid >> 5;
    const int wm   = wave & 3;
    const int wn   = wave >> 2;
    const int mrow0 = blockIdx.x * 128;
    const int nrow0 = blockIdx.y * 64;
    const int lm = lane & 15;
    const int hf = lane >> 4;

    v8f c[2][2];
    for (int i = 0; i < 2; ++i)
        for (int j = 0; j < 2; ++j)
            c[i][j] = (v8f){0.f, 0.f, 0.f, 0.f, 0.f, 0.f, 0.f, 0.f};

    auto stage = [&](int k0, u16* buf) {
        stage_A(A, K, mrow0, k0, buf, tid);
        if (wave == 0)
            tdm_load_2d(ldsoff(buf + 128 * PADW),
                        Wt + (size_t)nrow0 * K + k0,
                        64u, (unsigned)N, (unsigned)K);
    };
    auto publish = [&]() {
        wait_async0();                                   // A chunks landed
        if (wave == 0) __builtin_amdgcn_s_wait_tensorcnt(0);  // B tile landed
        __syncthreads();
    };

    stage(0, lds[0]);
    publish();

    for (int k0 = 0; k0 < K; k0 += 32) {
        const int  buf  = (k0 >> 5) & 1;
        const bool more = (k0 + 32) < K;
        if (more) stage(k0 + 32, lds[buf ^ 1]);

        const u16* As = lds[buf];
        const u16* Bs = lds[buf] + 128 * PADW;
        FragB a[2], b[2];
        for (int i = 0; i < 2; ++i) {
            const u16* ap = As + (wm * 32 + i * 16 + lm) * PADW + 8 * hf;
            a[i].u[0] = ld16(ap);
            a[i].u[1] = ld16(ap + 16);
        }
        for (int j = 0; j < 2; ++j) {
            const u16* bp = Bs + (wn * 32 + j * 16 + lm) * PADW + 8 * hf;
            b[j].u[0] = ld16(bp);
            b[j].u[1] = ld16(bp + 16);
        }
        for (int i = 0; i < 2; ++i)
            for (int j = 0; j < 2; ++j)
                c[i][j] = wmma_bf16(a[i].bf, b[j].bf, c[i][j]);

        if (more) publish();
    }

    for (int j = 0; j < 2; ++j) {
        const int   col  = nrow0 + wn * 32 + j * 16 + lm;
        const float bval = bias ? bias[col] : 0.f;
        for (int i = 0; i < 2; ++i) {
            for (int r = 0; r < 8; ++r) {
                int   row = mrow0 + wm * 32 + i * 16 + r + 8 * hf;
                float v   = c[i][j][r] + bval;
                if (RELU) v = fmaxf(v, 0.f);
                if (OUT_BF16) Cb[(size_t)row * N + col] = f2b(v);
                else          Cf[(size_t)row * N + col] = v;
            }
        }
    }
}

// ---------------------------------------------------------------------------
// Fused attention: one wave per (16-query tile, b, h).
// scores = Q K^T / 8, keep only strictly-upper (j > i) per the reference's
// inverted mask, fp32 softmax in LDS, then O = P @ V against Vt [B,H,DK,S].
// ---------------------------------------------------------------------------
__global__ __launch_bounds__(32)
void attn_kernel(const u16* __restrict__ QKV, const u16* __restrict__ Vt,
                 u16* __restrict__ Ob) {
    __shared__ float sm[16 * Ss];          // 64 KB score/prob tile

    const int lane = threadIdx.x;
    const int qt   = blockIdx.x;
    const int bh   = blockIdx.y;
    const int b    = bh / Hh;
    const int h    = bh % Hh;
    const int lm   = lane & 15;
    const int hf   = lane >> 4;
    const size_t qrow0 = (size_t)b * Ss + qt * 16;

    FragB qa[2];
    for (int kk = 0; kk < 2; ++kk) {
        const u16* qp = QKV + (qrow0 + lm) * QKVN + h * DKk + kk * 32 + 8 * hf;
        qa[kk].u[0] = ld16(qp);
        qa[kk].u[1] = ld16(qp + 16);
    }

    for (int j = 0; j < Ss / 16; ++j) {
        v8f c = (v8f){0.f, 0.f, 0.f, 0.f, 0.f, 0.f, 0.f, 0.f};
        for (int kk = 0; kk < 2; ++kk) {
            FragB kb;
            const u16* kp = QKV + ((size_t)b * Ss + j * 16 + lm) * QKVN + Dd +
                            h * DKk + kk * 32 + 8 * hf;
            kb.u[0] = ld16(kp);
            kb.u[1] = ld16(kp + 16);
            c = wmma_bf16(qa[kk].bf, kb.bf, c);
        }
        const int kcol = j * 16 + lm;
        for (int r = 0; r < 8; ++r) {
            int   rrow = r + 8 * hf;
            int   qrow = qt * 16 + rrow;
            float v    = c[r] * 0.125f;            // 1/sqrt(DK)
            if (kcol <= qrow) v = NEGBIG;          // faithful inverted mask
            sm[rrow * Ss + kcol] = v;
        }
    }
    __syncthreads();

    for (int r = 0; r < 16; ++r) {
        float m = -3.4e38f;
        for (int idx = lane; idx < Ss; idx += 32) m = fmaxf(m, sm[r * Ss + idx]);
        for (int o = 16; o > 0; o >>= 1) m = fmaxf(m, __shfl_xor(m, o, 32));
        float ssum = 0.f;
        for (int idx = lane; idx < Ss; idx += 32) {
            float e = __expf(sm[r * Ss + idx] - m);
            sm[r * Ss + idx] = e;
            ssum += e;
        }
        for (int o = 16; o > 0; o >>= 1) ssum += __shfl_xor(ssum, o, 32);
        float inv = 1.f / ssum;
        for (int idx = lane; idx < Ss; idx += 32) sm[r * Ss + idx] *= inv;
    }
    __syncthreads();

    v8f o[4];
    for (int nt = 0; nt < 4; ++nt)
        o[nt] = (v8f){0.f, 0.f, 0.f, 0.f, 0.f, 0.f, 0.f, 0.f};

    for (int j = 0; j < Ss / 32; ++j) {
        FragB pa;
        for (int e = 0; e < 8; ++e)
            pa.us[e]     = f2b(sm[lm * Ss + j * 32 + 8 * hf + e]);
        for (int e = 0; e < 8; ++e)
            pa.us[8 + e] = f2b(sm[lm * Ss + j * 32 + 16 + 8 * hf + e]);
        for (int nt = 0; nt < 4; ++nt) {
            FragB vb;
            const u16* vp = Vt + ((size_t)bh * DKk + nt * 16 + lm) * Ss +
                            (size_t)j * 32 + 8 * hf;
            vb.u[0] = ld16(vp);
            vb.u[1] = ld16(vp + 16);
            o[nt] = wmma_bf16(pa.bf, vb.bf, o[nt]);
        }
    }

    for (int nt = 0; nt < 4; ++nt) {
        const int col = h * DKk + nt * 16 + lm;
        for (int r = 0; r < 8; ++r) {
            int row = qt * 16 + r + 8 * hf;
            Ob[((size_t)b * Ss + row) * Dd + col] = f2b(o[nt][r]);
        }
    }
}

// ---------------------------------------------------------------------------
// Residual + LayerNorm: h = LN(X + Delta) * g + be  (writes f32 + bf16)
// ---------------------------------------------------------------------------
__global__ __launch_bounds__(256)
void ln_kernel(const float* __restrict__ X, const float* __restrict__ Dl,
               const float* __restrict__ g, const float* __restrict__ be,
               float* __restrict__ Hf, u16* __restrict__ Hb) {
    __shared__ float red[256];
    __shared__ float stat[2];
    const int t   = blockIdx.x;
    const int tid = threadIdx.x;
    const size_t base = (size_t)t * Dd;

    float a0 = X[base + tid]       + Dl[base + tid];
    float a1 = X[base + 256 + tid] + Dl[base + 256 + tid];

    red[tid] = a0 + a1;
    __syncthreads();
    for (int s = 128; s > 0; s >>= 1) {
        if (tid < s) red[tid] += red[tid + s];
        __syncthreads();
    }
    if (tid == 0) stat[0] = red[0] * (1.f / Dd);
    __syncthreads();
    const float mu = stat[0];

    float d0 = a0 - mu, d1 = a1 - mu;
    red[tid] = d0 * d0 + d1 * d1;
    __syncthreads();
    for (int s = 128; s > 0; s >>= 1) {
        if (tid < s) red[tid] += red[tid + s];
        __syncthreads();
    }
    if (tid == 0) stat[1] = rsqrtf(red[0] * (1.f / Dd) + 1e-5f);
    __syncthreads();
    const float rstd = stat[1];

    float y0 = d0 * rstd * g[tid]       + be[tid];
    float y1 = d1 * rstd * g[256 + tid] + be[256 + tid];
    Hf[base + tid]       = y0;  Hb[base + tid]       = f2b(y0);
    Hf[base + 256 + tid] = y1;  Hb[base + 256 + tid] = f2b(y1);
}

// ---------------------------------------------------------------------------
// Host-side orchestration
// ---------------------------------------------------------------------------
extern "C" void kernel_launch(void* const* d_in, const int* in_sizes, int n_in,
                              void* d_out, int out_size, void* d_ws, size_t ws_size,
                              hipStream_t stream) {
    (void)in_sizes; (void)n_in; (void)out_size; (void)ws_size;

    const int*   ids  = (const int*)  d_in[0];
    const float* emb  = (const float*)d_in[1];
    const float* pe   = (const float*)d_in[2];
    const float* Wq   = (const float*)d_in[3];
    const float* bq   = (const float*)d_in[4];
    const float* Wk   = (const float*)d_in[5];
    const float* bk   = (const float*)d_in[6];
    const float* Wv   = (const float*)d_in[7];
    const float* bv   = (const float*)d_in[8];
    const float* Wo   = (const float*)d_in[9];
    const float* bo   = (const float*)d_in[10];
    const float* ln1g = (const float*)d_in[11];
    const float* ln1b = (const float*)d_in[12];
    const float* W1   = (const float*)d_in[13];
    const float* b1   = (const float*)d_in[14];
    const float* W2   = (const float*)d_in[15];
    const float* b2   = (const float*)d_in[16];
    const float* ln2g = (const float*)d_in[17];
    const float* ln2b = (const float*)d_in[18];
    const float* Wout = (const float*)d_in[19];
    const float* bout = (const float*)d_in[20];
    float* logits = (float*)d_out;

    char*  wsb = (char*)d_ws;
    size_t off = 0;
    auto carve = [&](size_t bytes) -> char* {
        char* p = wsb + off;
        off += (bytes + 255) & ~(size_t)255;
        return p;
    };
    u16*   wqkv_b = (u16*)  carve((size_t)Ll * Dd * QKVN * 2);   // [L][1536][512]
    u16*   wo_b   = (u16*)  carve((size_t)Ll * Dd * Dd * 2);
    u16*   w1_b   = (u16*)  carve((size_t)Ll * Dd * Ff * 2);
    u16*   w2_b   = (u16*)  carve((size_t)Ll * Ff * Dd * 2);
    u16*   wout_b = (u16*)  carve((size_t)Dd * Vv * 2);
    float* bqkv   = (float*)carve((size_t)Ll * QKVN * 4);
    float* h_f    = (float*)carve((size_t)MTOK * Dd * 4);
    u16*   h_b    = (u16*)  carve((size_t)MTOK * Dd * 2);
    u16*   qkv_b  = (u16*)  carve((size_t)MTOK * QKVN * 2);
    u16*   vt_b   = (u16*)  carve((size_t)MTOK * Dd * 2);
    u16*   ao_b   = (u16*)  carve((size_t)MTOK * Dd * 2);
    float* del_f  = (float*)carve((size_t)MTOK * Dd * 4);
    u16*   mid_b  = (u16*)  carve((size_t)MTOK * Ff * 2);

    const long DD = (long)Dd * Dd;
    auto launch_convT = [&](const float* src, u16* dst, int K, int N, long total,
                            long layerStride, long base) {
        int blocks = (int)((total + 255) / 256);
        convT_kernel<<<blocks, 256, 0, stream>>>(src, dst, K, N, total,
                                                 layerStride, base);
    };
    launch_convT(Wq,   wqkv_b, Dd, Dd, Ll * DD,            3 * DD,          0);
    launch_convT(Wk,   wqkv_b, Dd, Dd, Ll * DD,            3 * DD,          DD);
    launch_convT(Wv,   wqkv_b, Dd, Dd, Ll * DD,            3 * DD,          2 * DD);
    launch_convT(Wo,   wo_b,   Dd, Dd, Ll * DD,            DD,              0);
    launch_convT(W1,   w1_b,   Dd, Ff, (long)Ll * Dd * Ff, (long)Dd * Ff,   0);
    launch_convT(W2,   w2_b,   Ff, Dd, (long)Ll * Ff * Dd, (long)Ff * Dd,   0);
    launch_convT(Wout, wout_b, Dd, Vv, (long)Dd * Vv,      (long)Dd * Vv,   0);
    biascat_kernel<<<(Ll * QKVN + 255) / 256, 256, 0, stream>>>(bq, bk, bv, bqkv);

    embed_kernel<<<MTOK, 256, 0, stream>>>(ids, emb, pe, h_f, h_b);

    const dim3 gQKV(MTOK / 128, QKVN / 64);   // N=1536
    const dim3 gD  (MTOK / 128, Dd / 64);     // N=512
    const dim3 gF  (MTOK / 128, Ff / 64);     // N=2048
    const dim3 gV  (MTOK / 128, Vv / 64);     // N=512 (output proj)

    for (int l = 0; l < Ll; ++l) {
        const size_t wQKV = (size_t)l * Dd * QKVN;
        const size_t wDD  = (size_t)l * Dd * Dd;
        const size_t wDF  = (size_t)l * Dd * Ff;

        gemm_kernel<true,  false><<<gQKV, 256, 0, stream>>>(
            h_b, wqkv_b + wQKV, bqkv + (size_t)l * QKVN, nullptr, qkv_b,
            MTOK, QKVN, Dd);

        vtrans_kernel<<<(MTOK * Dd) / 256, 256, 0, stream>>>(qkv_b, vt_b);

        attn_kernel<<<dim3(Ss / 16, Bb * Hh), 32, 0, stream>>>(qkv_b, vt_b, ao_b);

        gemm_kernel<false, false><<<gD, 256, 0, stream>>>(
            ao_b, wo_b + wDD, bo + l * Dd, del_f, nullptr, MTOK, Dd, Dd);
        ln_kernel<<<MTOK, 256, 0, stream>>>(h_f, del_f, ln1g + l * Dd,
                                            ln1b + l * Dd, h_f, h_b);

        gemm_kernel<true,  true ><<<gF, 256, 0, stream>>>(
            h_b, w1_b + wDF, b1 + l * Ff, nullptr, mid_b, MTOK, Ff, Dd);
        gemm_kernel<false, false><<<gD, 256, 0, stream>>>(
            mid_b, w2_b + wDF, b2 + l * Dd, del_f, nullptr, MTOK, Dd, Ff);
        ln_kernel<<<MTOK, 256, 0, stream>>>(h_f, del_f, ln2g + l * Dd,
                                            ln2b + l * Dd, h_f, h_b);
    }

    gemm_kernel<false, false><<<gV, 256, 0, stream>>>(
        h_b, wout_b, bout, logits, nullptr, MTOK, Vv, Dd);
}
